// CasualMHSA_63900523430331
// MI455X (gfx1250) — matrix operand order repaired and checked by reading, not compiled, not run on hardware
//
#include <hip/hip_runtime.h>
#include <hip/hip_bf16.h>
#include <cstdint>

typedef __attribute__((ext_vector_type(16))) __bf16 v16bf;
typedef __attribute__((ext_vector_type(8)))  float  v8f;
typedef __attribute__((ext_vector_type(4)))  unsigned int v4u;
typedef __attribute__((ext_vector_type(8)))  int    v8i;
typedef __attribute__((ext_vector_type(4)))  int    v4i;

#define WMMA_BF16(a, b, c) \
  __builtin_amdgcn_wmma_f32_16x16x32_bf16(false, (a), false, (b), (short)0, (c), false, false)

static constexpr int D_MODEL = 1024;
static constexpr int N_HEAD  = 16;
static constexpr int HD      = 64;
static constexpr int BATCH   = 2;
static constexpr int SEQ     = 2048;
static constexpr int M_ROWS  = BATCH * SEQ;   // 4096

__device__ __forceinline__ v8f vzero8() {
  v8f z = {0.0f, 0.0f, 0.0f, 0.0f, 0.0f, 0.0f, 0.0f, 0.0f};
  return z;
}

// ---- CDNA5 async copy: global -> LDS, 16B per lane, ASYNCcnt-tracked ------
__device__ __forceinline__ uint32_t lds_off(const void* p) {
  // flat LDS pointers carry the wave-relative LDS address in addr[31:0]
  return (uint32_t)(uintptr_t)p;
}
__device__ __forceinline__ void async_b128(uint32_t lds, uint32_t voff, uint64_t sbase) {
  asm volatile("global_load_async_to_lds_b128 %0, %1, %2"
               :: "v"(lds), "v"(voff), "s"(sbase) : "memory");
}
__device__ __forceinline__ void wait_async_le4() {
  asm volatile("s_wait_asynccnt 0x4" ::: "memory");
}
__device__ __forceinline__ void wait_async_0() {
  asm volatile("s_wait_asynccnt 0x0" ::: "memory");
}

// ---- CDNA5 Tensor Data Mover: 2D bf16 tile (64x64) -> LDS with padding ----
// D# per CDNA5 ISA. data_size=2B; LDS padding of
// 4 DWORDs after every 32 DWORDs -> destination row stride of 72 elements.
// This toolchain exposes the 6-arg builtin: (v4u, v8i, v4i, v4i, v8i, i32).
__device__ __forceinline__ void tdm_load_tile_64x64(uint32_t lds, uint64_t gaddr,
                                                    uint32_t tdim0, uint32_t tdim1,
                                                    uint32_t stride0) {
  v4u g0 = { 1u,                                              // count=1, user D#
             lds,                                             // lds_addr
             (uint32_t)(gaddr & 0xffffffffu),                 // global_addr[31:0]
             (uint32_t)((gaddr >> 32) & 0x1ffffffu) | (2u << 30) }; // [56:32] | type=2
  const uint32_t w0 = (1u << 16)    // data_size: 2 bytes
                    | (1u << 20)    // pad_enable
                    | (4u << 22)    // pad_interval: 32 DWORDs (1<<4 x 8B)
                    | (3u << 25);   // pad_amount: 4 DWORDs
  v8i g1 = { (int)w0,
             (int)((tdim0 & 0xffffu) << 16),                       // tensor_dim0[15:0]
             (int)((tdim0 >> 16) | ((tdim1 & 0xffffu) << 16)),     // dim0 hi | dim1 lo
             (int)((tdim1 >> 16) | (64u << 16)),                   // dim1 hi | tile_dim0=64
             (int)64,                                              // tile_dim1=64, tile_dim2=0
             (int)stride0,                                         // tensor_dim0_stride lo
             0, 0 };
  v4i gz4 = { 0, 0, 0, 0 };
  v8i gz8 = { 0, 0, 0, 0, 0, 0, 0, 0 };
  __builtin_amdgcn_tensor_load_to_lds(g0, g1, gz4, gz4, gz8, 0);
}

// ---------------------------------------------------------------------------
// f32 -> bf16 elementwise (8 elems / thread)
// ---------------------------------------------------------------------------
__global__ __launch_bounds__(256)
void cvt_bf16(const float* __restrict__ src, __bf16* __restrict__ dst, const int n)
{
  const int i = (blockIdx.x * 256 + threadIdx.x) * 8;
  if (i >= n) return;
  const float4 a = ((const float4*)(src + i))[0];
  const float4 b = ((const float4*)(src + i))[1];
  __align__(16) __bf16 h[8];
  h[0] = (__bf16)a.x; h[1] = (__bf16)a.y; h[2] = (__bf16)a.z; h[3] = (__bf16)a.w;
  h[4] = (__bf16)b.x; h[5] = (__bf16)b.y; h[6] = (__bf16)b.z; h[7] = (__bf16)b.w;
  *(uint4*)(dst + i) = *(uint4*)h;
}

// ---------------------------------------------------------------------------
// GEMM: Y[m,n] = sum_k A[m,k] * W[n,k] + bias[n]; A,W bf16 row-major [.,K].
// Double-buffered LDS tiles filled by GLOBAL_LOAD_ASYNC_TO_LDS_B128.
// mode 0: scatter Q(x1/8)/K into [B,H,T,64], V into [B,H,64,T] (transposed).
// mode 1: f32 out[m*1024+n].
// ---------------------------------------------------------------------------
__global__ __launch_bounds__(256)
void gemm_wmma(const __bf16* __restrict__ A, const __bf16* __restrict__ W,
               const float* __restrict__ bias, const int K, const int mode,
               __bf16* __restrict__ qout, __bf16* __restrict__ kout,
               __bf16* __restrict__ vout, float* __restrict__ out)
{
  __shared__ __bf16 As[2][128][32];
  __shared__ __bf16 Bs[2][128][32];

  const int tid  = threadIdx.x;
  const int lane = tid & 31;
  const int wave = tid >> 5;
  const int wm   = wave & 3;
  const int wn   = wave >> 2;
  const int g    = lane >> 4;
  const int l16  = lane & 15;

  const int m0 = blockIdx.x * 128;
  const int n0 = blockIdx.y * 128;

  v8f acc[2][4];
#pragma unroll
  for (int mt = 0; mt < 2; ++mt)
#pragma unroll
    for (int nt = 0; nt < 4; ++nt) acc[mt][nt] = vzero8();

  const int lr = tid >> 1;
  const int lc = (tid & 1) * 16;
  const uint64_t abase = (uint64_t)(uintptr_t)A;
  const uint64_t wbase = (uint64_t)(uintptr_t)W;
  uint32_t aoff = (uint32_t)(((m0 + lr) * K + lc) * 2);
  uint32_t woff = (uint32_t)(((n0 + lr) * K + lc) * 2);
  const uint32_t ldsA[2] = { lds_off(&As[0][lr][lc]), lds_off(&As[1][lr][lc]) };
  const uint32_t ldsB[2] = { lds_off(&Bs[0][lr][lc]), lds_off(&Bs[1][lr][lc]) };

  async_b128(ldsA[0],      aoff,      abase);
  async_b128(ldsA[0] + 16, aoff + 16, abase);
  async_b128(ldsB[0],      woff,      wbase);
  async_b128(ldsB[0] + 16, woff + 16, wbase);
  aoff += 64; woff += 64;

  const int NK = K >> 5;
  for (int i = 0; i < NK; ++i) {
    const int buf = i & 1;
    if (i + 1 < NK) {
      const int nb = buf ^ 1;
      async_b128(ldsA[nb],      aoff,      abase);
      async_b128(ldsA[nb] + 16, aoff + 16, abase);
      async_b128(ldsB[nb],      woff,      wbase);
      async_b128(ldsB[nb] + 16, woff + 16, wbase);
      aoff += 64; woff += 64;
      wait_async_le4();
    } else {
      wait_async_0();
    }
    __syncthreads();

    v16bf af[2];
#pragma unroll
    for (int mt = 0; mt < 2; ++mt) {
      const __bf16* ap = &As[buf][wm * 32 + mt * 16 + l16][g * 8];
      ((uint4*)&af[mt])[0] = *(const uint4*)ap;
      ((uint4*)&af[mt])[1] = *(const uint4*)(ap + 16);
    }
    v16bf wf[4];
#pragma unroll
    for (int nt = 0; nt < 4; ++nt) {
      const __bf16* bp = &Bs[buf][wn * 64 + nt * 16 + l16][g * 8];
      ((uint4*)&wf[nt])[0] = *(const uint4*)bp;
      ((uint4*)&wf[nt])[1] = *(const uint4*)(bp + 16);
    }
#pragma unroll
    for (int mt = 0; mt < 2; ++mt)
#pragma unroll
      for (int nt = 0; nt < 4; ++nt)
        acc[mt][nt] = WMMA_BF16(af[mt], wf[nt], acc[mt][nt]);
    __syncthreads();
  }

#pragma unroll
  for (int mt = 0; mt < 2; ++mt) {
#pragma unroll
    for (int nt = 0; nt < 4; ++nt) {
      const int n  = n0 + wn * 64 + nt * 16 + l16;
      const float bv = bias[n];
#pragma unroll
      for (int r = 0; r < 8; ++r) {
        const int m = m0 + wm * 32 + mt * 16 + r + 8 * g;
        const float val = acc[mt][nt][r] + bv;
        if (mode == 0) {
          const int which = n >> 10;        // 0=q 1=k 2=v
          const int d  = n & 1023;
          const int h  = d >> 6;
          const int hi = d & 63;
          const int bb = m >> 11;
          const int t  = m & 2047;
          if (which == 0) {
            qout[((size_t)(bb * N_HEAD + h) * SEQ + t) * HD + hi] =
                (__bf16)(val * 0.125f);     // 1/sqrt(64)
          } else if (which == 1) {
            kout[((size_t)(bb * N_HEAD + h) * SEQ + t) * HD + hi] = (__bf16)val;
          } else {                          // V stored transposed: [B,H,hd,T]
            vout[((size_t)(bb * N_HEAD + h) * HD + hi) * SEQ + t] = (__bf16)val;
          }
        } else {
          out[(size_t)m * D_MODEL + n] = val;
        }
      }
    }
  }
}

// ---------------------------------------------------------------------------
// Flash attention, causal. Grid (T/128, B*H), 256 thr = 8 waves.
// K tiles [key][hd] and pre-transposed V tiles [hd][key] are DMA'd into
// double-buffered LDS by the Tensor Data Mover (wave 0 issues, TENSORcnt
// waits, barrier publishes). TDM LDS padding gives the 72-elem row stride.
// ---------------------------------------------------------------------------
__global__ __launch_bounds__(256)
void flash_attn(const __bf16* __restrict__ Qg, const __bf16* __restrict__ Kg,
                const __bf16* __restrict__ Vtg, __bf16* __restrict__ ctx)
{
  __shared__ __bf16 Ks[2][64][72];
  __shared__ __bf16 Vt[2][64][72];
  __shared__ __bf16 Ps[8][16][72];

  const int tid  = threadIdx.x;
  const int lane = tid & 31;
  const int wave = tid >> 5;
  const int g    = lane >> 4;
  const int l16  = lane & 15;

  const int q0 = blockIdx.x * 128;
  const int bh = blockIdx.y;
  const size_t base = (size_t)bh * SEQ * HD;   // head base (K and V^T equal-sized)
  const int qw0 = q0 + wave * 16;

  v16bf qa[2];
  {
    const __bf16* qp = Qg + base + (size_t)(qw0 + l16) * HD;
#pragma unroll
    for (int c = 0; c < 2; ++c) {
      ((uint4*)&qa[c])[0] = *(const uint4*)(qp + c * 32 + g * 8);
      ((uint4*)&qa[c])[1] = *(const uint4*)(qp + c * 32 + 16 + g * 8);
    }
  }

  v8f o[4];
#pragma unroll
  for (int ht = 0; ht < 4; ++ht) o[ht] = vzero8();
  float mrow[8], lrow[8];
#pragma unroll
  for (int r = 0; r < 8; ++r) { mrow[r] = -1e30f; lrow[r] = 0.0f; }

  const uint64_t kgaddr = (uint64_t)(uintptr_t)(Kg + base);
  const uint64_t vgaddr = (uint64_t)(uintptr_t)(Vtg + base);
  const uint32_t ldsKb[2] = { lds_off(&Ks[0][0][0]), lds_off(&Ks[1][0][0]) };
  const uint32_t ldsVb[2] = { lds_off(&Vt[0][0][0]), lds_off(&Vt[1][0][0]) };

  const int kv_end = q0 + 128;
  const int niter  = kv_end >> 6;

  // prologue: kv tile 0 -> buffer 0 (TDM, wave 0 only; EXEC-independent DMA)
  if (wave == 0) {
    // K head: tensor [T=2048 rows x 64], tile rows kv0..kv0+63
    tdm_load_tile_64x64(ldsKb[0], kgaddr, HD, SEQ, HD);
    // V^T head: tensor [64 rows x T=2048], tile cols kv0..kv0+63
    tdm_load_tile_64x64(ldsVb[0], vgaddr, SEQ, HD, SEQ);
  }

  for (int i = 0; i < niter; ++i) {
    const int buf = i & 1;
    const int kv0 = i * 64;
    if (wave == 0) {
      if (i + 1 < niter) {
        const int nb = buf ^ 1;
        const uint64_t kv1 = (uint64_t)(kv0 + 64);
        tdm_load_tile_64x64(ldsKb[nb], kgaddr + kv1 * HD * 2, HD, SEQ, HD);
        tdm_load_tile_64x64(ldsVb[nb], vgaddr + kv1 * 2,      SEQ, HD, SEQ);
        __builtin_amdgcn_s_wait_tensorcnt((short)2);  // current tile's 2 DMAs done
      } else {
        __builtin_amdgcn_s_wait_tensorcnt((short)0);
      }
    }
    __syncthreads();

    if (kv0 <= qw0 + 15) {   // wave-uniform causal skip
      v8f s[4];
#pragma unroll
      for (int nt = 0; nt < 4; ++nt) {
        s[nt] = vzero8();
#pragma unroll
        for (int c = 0; c < 2; ++c) {
          v16bf kf;
          const __bf16* kl = &Ks[buf][nt * 16 + l16][c * 32 + g * 8];
          ((uint4*)&kf)[0] = *(const uint4*)kl;
          ((uint4*)&kf)[1] = *(const uint4*)(kl + 16);
          s[nt] = WMMA_BF16(qa[c], kf, s[nt]);
        }
      }
#pragma unroll
      for (int r = 0; r < 8; ++r) {
        const int qrow = qw0 + r + 8 * g;
        float tmax = -1e30f;
#pragma unroll
        for (int nt = 0; nt < 4; ++nt) {
          const int kc = kv0 + nt * 16 + l16;
          float sv = s[nt][r];
          if (kc > qrow) { sv = -1e30f; s[nt][r] = sv; }
          tmax = fmaxf(tmax, sv);
        }
#pragma unroll
        for (int off = 1; off < 16; off <<= 1)
          tmax = fmaxf(tmax, __shfl_xor(tmax, off, 16));
        const float mnew  = fmaxf(mrow[r], tmax);
        const float alpha = __expf(mrow[r] - mnew);
        mrow[r] = mnew;
        float psum = 0.0f;
#pragma unroll
        for (int nt = 0; nt < 4; ++nt) {
          const float p = __expf(s[nt][r] - mnew);
          s[nt][r] = p;
          psum += p;
        }
#pragma unroll
        for (int off = 1; off < 16; off <<= 1)
          psum += __shfl_xor(psum, off, 16);
        lrow[r] = lrow[r] * alpha + psum;
#pragma unroll
        for (int ht = 0; ht < 4; ++ht) o[ht][r] *= alpha;
      }
#pragma unroll
      for (int nt = 0; nt < 4; ++nt)
#pragma unroll
        for (int r = 0; r < 8; ++r)
          Ps[wave][r + 8 * g][nt * 16 + l16] = (__bf16)s[nt][r];
#pragma unroll
      for (int c = 0; c < 2; ++c) {
        v16bf pf;
        const __bf16* pp = &Ps[wave][l16][c * 32 + g * 8];
        ((uint4*)&pf)[0] = *(const uint4*)pp;
        ((uint4*)&pf)[1] = *(const uint4*)(pp + 16);
#pragma unroll
        for (int ht = 0; ht < 4; ++ht) {
          v16bf vf;
          const __bf16* vl = &Vt[buf][ht * 16 + l16][c * 32 + g * 8];
          ((uint4*)&vf)[0] = *(const uint4*)vl;
          ((uint4*)&vf)[1] = *(const uint4*)(vl + 16);
          o[ht] = WMMA_BF16(pf, vf, o[ht]);
        }
      }
    }
    __syncthreads();
  }

  const int b = bh >> 4;
  const int h = bh & 15;
#pragma unroll
  for (int r = 0; r < 8; ++r) {
    const float inv = 1.0f / lrow[r];
    const int qrow = qw0 + r + 8 * g;
    const size_t row = (size_t)(b * SEQ + qrow) * D_MODEL;
#pragma unroll
    for (int ht = 0; ht < 4; ++ht) {
      const int col = h * HD + ht * 16 + l16;
      ctx[row + col] = (__bf16)(o[ht][r] * inv);
    }
  }
}

// ---------------------------------------------------------------------------
extern "C" void kernel_launch(void* const* d_in, const int* in_sizes, int n_in,
                              void* d_out, int out_size, void* d_ws, size_t ws_size,
                              hipStream_t stream)
{
  (void)in_sizes; (void)n_in; (void)out_size; (void)ws_size;
  const float* x     = (const float*)d_in[0];
  const float* w_in  = (const float*)d_in[1];
  const float* b_in  = (const float*)d_in[2];
  const float* w_out = (const float*)d_in[3];
  const float* b_out = (const float*)d_in[4];
  float* out = (float*)d_out;

  char* ws = (char*)d_ws;
  const size_t MB = 1024 * 1024;
  __bf16* xh   = (__bf16*)(ws);             //  8 MB  x bf16 [4096,1024]
  __bf16* wih  = (__bf16*)(ws + 8  * MB);   //  6 MB  w_in bf16 [3072,1024]
  __bf16* woh  = (__bf16*)(ws + 14 * MB);   //  2 MB  w_out bf16 [1024,1024]
  __bf16* qb   = (__bf16*)(ws + 16 * MB);   //  8 MB  Q  [B,H,T,64]
  __bf16* kb   = (__bf16*)(ws + 24 * MB);   //  8 MB  K  [B,H,T,64]
  __bf16* vb   = (__bf16*)(ws + 32 * MB);   //  8 MB  V^T[B,H,64,T]
  __bf16* ctxb = (__bf16*)(ws + 40 * MB);   //  8 MB  ctx [4096,1024]

  dim3 blk(256);
  const int nx = M_ROWS * D_MODEL, nwi = 3 * D_MODEL * D_MODEL, nwo = D_MODEL * D_MODEL;
  cvt_bf16<<<nx  / 2048, blk, 0, stream>>>(x,     xh,  nx);
  cvt_bf16<<<nwi / 2048, blk, 0, stream>>>(w_in,  wih, nwi);
  cvt_bf16<<<nwo / 2048, blk, 0, stream>>>(w_out, woh, nwo);

  dim3 g1(M_ROWS / 128, (3 * D_MODEL) / 128);   // 32 x 24
  gemm_wmma<<<g1, blk, 0, stream>>>(xh, wih, b_in, D_MODEL, 0, qb, kb, vb, nullptr);

  dim3 g2(SEQ / 128, BATCH * N_HEAD);           // 16 x 32
  flash_attn<<<g2, blk, 0, stream>>>(qb, kb, vb, ctxb);

  dim3 g3(M_ROWS / 128, D_MODEL / 128);         // 32 x 8
  gemm_wmma<<<g3, blk, 0, stream>>>(ctxb, woh, b_out, D_MODEL, 1,
                                    nullptr, nullptr, nullptr, out);
}
